// SelfAttention_37452114821624
// MI455X (gfx1250) — compile-verified, matrix operand order checked by reading, not run on hardware
//
#include <hip/hip_runtime.h>
#include <hip/hip_bf16.h>

typedef __attribute__((ext_vector_type(16))) _Float16 v16h;
typedef __attribute__((ext_vector_type(8)))  _Float16 v8h;
typedef __attribute__((ext_vector_type(4)))  _Float16 v4h;
typedef __attribute__((ext_vector_type(8)))  float    v8f;
typedef __attribute__((ext_vector_type(4)))  int      v4i;
typedef __attribute__((ext_vector_type(4)))  unsigned u32x4;
typedef __attribute__((ext_vector_type(8)))  int      i32x8;
typedef __attribute__((ext_vector_type(4)))  int      i32x4;

#define D_MODEL   1024
#define SEQ       2048
#define BATCH     4
#define MTOT      (BATCH * SEQ)     // 8192 rows
#define QS_STRIDE 1032              // 1024 + 8 halves pad
#define VT_STRIDE 40                // 32 + 8 halves pad (matches TDM pad config)
#define PS_STRIDE 40

// ---------------------------------------------------------------------------
// CDNA5 async global->LDS copy (ASYNCcnt), with sync fallback.
// Probe-verified signature: (v4i AS1*, v4i AS3*, imm offset, imm cpol).
// ---------------------------------------------------------------------------
#if defined(__has_builtin)
#if __has_builtin(__builtin_amdgcn_global_load_async_to_lds_b128) && \
    __has_builtin(__builtin_amdgcn_s_wait_asynccnt)
#define HAVE_ASYNC_LDS 1
#endif
#if __has_builtin(__builtin_amdgcn_tensor_load_to_lds) && \
    __has_builtin(__builtin_amdgcn_s_wait_tensorcnt)
#define HAVE_TDM 1
#endif
#endif

#ifdef HAVE_ASYNC_LDS
__device__ __forceinline__ void cp_b128(void* lds, const void* g) {
    __builtin_amdgcn_global_load_async_to_lds_b128(
        (__attribute__((address_space(1))) v4i*)(void*)g,
        (__attribute__((address_space(3))) v4i*)lds, 0, 0);
}
__device__ __forceinline__ void wait_async() { __builtin_amdgcn_s_wait_asynccnt(0); }
#else
__device__ __forceinline__ void cp_b128(void* lds, const void* g) {
    *(v8h*)lds = *(const v8h*)g;
}
__device__ __forceinline__ void wait_async() {}
#endif

#ifdef HAVE_TDM
__device__ __forceinline__ unsigned lds_addr_of(void* p) {
    return (unsigned)(unsigned long long)(__attribute__((address_space(3))) char*)p;
}
// 2-D TENSOR_LOAD_TO_LDS: D# per CDNA5 ISA §8.3/8.4 (16-bit elements).
// pad_interval enum e -> pad every 2^(e+1) DWORDs; pad_amount enum a -> a+1 DWORDs.
// This toolchain carries the 6-arg builtin: (u32x4, i32x8, i32x4, i32x4, i32x8, i32).
__device__ __forceinline__ void tdm_load_2d(
    unsigned lds, unsigned long long gaddr,
    unsigned tile_w, unsigned tile_h,          // tile dims (elements)
    unsigned tens_w, unsigned tens_h,          // tensor dims (elements)
    unsigned long long row_stride,             // tensor dim0 stride (elements)
    unsigned pad_interval, unsigned pad_amount, unsigned pad_en)
{
    u32x4 g0;
    g0[0] = 1u;                                               // count=1, user D#
    g0[1] = lds;                                              // lds_addr (bytes)
    g0[2] = (unsigned)(gaddr & 0xffffffffu);                  // global_addr[31:0]
    g0[3] = (unsigned)((gaddr >> 32) & 0x1ffffffu) | (2u << 30); // ga[56:32] | type=2
    i32x8 g1;
    g1[0] = (int)((1u << 16) | (pad_en << 20) | (pad_interval << 22) | (pad_amount << 25));
    g1[1] = (int)((tens_w & 0xffffu) << 16);                  // tensor_dim0[15:0]
    g1[2] = (int)((tens_w >> 16) | ((tens_h & 0xffffu) << 16)); // dim0[31:16]|dim1[15:0]
    g1[3] = (int)((tens_h >> 16) | (tile_w << 16));           // dim1[31:16]|tile_dim0
    g1[4] = (int)(tile_h & 0xffffu);                          // tile_dim1 (tile_dim2=0)
    g1[5] = (int)(row_stride & 0xffffffffu);                  // dim0_stride[31:0]
    g1[6] = (int)((row_stride >> 32) & 0xffffu);              // dim0_stride[47:32]
    g1[7] = 0;
    i32x4 z4 = {0, 0, 0, 0};
    i32x8 z8 = {0, 0, 0, 0, 0, 0, 0, 0};
    __builtin_amdgcn_tensor_load_to_lds(g0, g1, z4, z4, z8, 0);
}
__device__ __forceinline__ void wait_tensor() { __builtin_amdgcn_s_wait_tensorcnt(0); }
#endif

// ---------------------------------------------------------------------------
// 16-bit WMMA A (16x32) / B (32x16) fragment loader (LDS or global).
// Lanes 0-15: K in [0..7]+[16..23]; lanes 16-31: K in [8..15]+[24..31].
// ---------------------------------------------------------------------------
__device__ __forceinline__ v16h ld_frag(const _Float16* base, int row, int stride, int k0) {
    const int lane = threadIdx.x & 31;
    const _Float16* p = base + row * stride + k0 + ((lane >> 4) & 1) * 8;
    v8h lo = *(const v8h*)(p);
    v8h hi = *(const v8h*)(p + 16);
    v16h r;
#pragma unroll
    for (int i = 0; i < 8; ++i) { r[i] = lo[i]; r[i + 8] = hi[i]; }
    return r;
}

// ---------------------------------------------------------------------------
// Kernel 1: fp32 -> fp16 cast
// ---------------------------------------------------------------------------
__global__ void cast_f32_to_f16(const float* __restrict__ in, _Float16* __restrict__ out, int n) {
    int i = (blockIdx.x * blockDim.x + threadIdx.x) * 4;
    if (i + 3 < n) {
        float4 v = *(const float4*)(in + i);
        v4h o;
        o[0] = (_Float16)v.x; o[1] = (_Float16)v.y; o[2] = (_Float16)v.z; o[3] = (_Float16)v.w;
        *(v4h*)(out + i) = o;
    }
}

// ---------------------------------------------------------------------------
// Kernel 2: fused QKV projection, double-buffered async LDS staging.
// WG tile 64(M) x 128(N); V output written transposed per batch: Vt[b][d][n].
// ---------------------------------------------------------------------------
__global__ void __launch_bounds__(256) qkv_gemm(
    const _Float16* __restrict__ xh,
    const _Float16* __restrict__ Wq, const _Float16* __restrict__ Wk, const _Float16* __restrict__ Wv,
    _Float16* __restrict__ Qh, _Float16* __restrict__ Kh, _Float16* __restrict__ Vtg)
{
    __shared__ _Float16 As[2][64 * 40];
    __shared__ _Float16 Bs[2][128 * 40];

    const int t = threadIdx.x, lane = t & 31, w = t >> 5;
    const int mbase = blockIdx.x * 64;
    const int yb = blockIdx.y;
    const int sel = yb >> 3;
    const int nbase = (yb & 7) * 128;
    const _Float16* W = (sel == 0) ? Wq : (sel == 1) ? Wk : Wv;

    const int mi = w & 3;
    const int nj = w >> 2;

    auto stage = [&](int buf, int kb) {
        {
            int h = t * 8, row = h >> 5, col = h & 31;
            cp_b128(&As[buf][row * 40 + col], xh + (size_t)(mbase + row) * D_MODEL + kb + col);
        }
#pragma unroll
        for (int j = 0; j < 2; ++j) {
            int h = t * 8 + j * 2048, row = h >> 5, col = h & 31;
            cp_b128(&Bs[buf][row * 40 + col], W + (size_t)(nbase + row) * D_MODEL + kb + col);
        }
    };

    v8f c[4];
#pragma unroll
    for (int s = 0; s < 4; ++s)
#pragma unroll
        for (int r = 0; r < 8; ++r) c[s][r] = 0.f;

    stage(0, 0);
    for (int kb = 0; kb < D_MODEL; kb += 32) {
        const int cur = (kb >> 5) & 1;
        wait_async();
        __syncthreads();
        if (kb + 32 < D_MODEL) stage(cur ^ 1, kb + 32);

        v16h a = ld_frag(As[cur], mi * 16 + (lane & 15), 40, 0);
#pragma unroll
        for (int s = 0; s < 4; ++s) {
            v16h b = ld_frag(Bs[cur], nj * 64 + s * 16 + (lane & 15), 40, 0);
            c[s] = __builtin_amdgcn_wmma_f32_16x16x32_f16(false, a, false, b, (short)0, c[s], false, false);
        }
    }

    const int rbase = mbase + mi * 16 + ((lane >> 4) ? 8 : 0);
    const int col0  = nbase + nj * 64 + (lane & 15);
    if (sel < 2) {
        _Float16* Out = sel ? Kh : Qh;
#pragma unroll
        for (int s = 0; s < 4; ++s)
#pragma unroll
            for (int r = 0; r < 8; ++r)
                Out[(size_t)(rbase + r) * D_MODEL + col0 + s * 16] = (_Float16)c[s][r];
    } else {
#pragma unroll
        for (int s = 0; s < 4; ++s)
#pragma unroll
            for (int r = 0; r < 8; ++r) {
                int m = rbase + r, bb = m >> 11, n = m & (SEQ - 1);
                int d = col0 + s * 16;
                Vtg[(size_t)bb * D_MODEL * SEQ + (size_t)d * SEQ + n] = (_Float16)c[s][r];
            }
    }
}

// ---------------------------------------------------------------------------
// Kernel 3: single-pass flash attention.
// K tiles: async b128 double-buffered.  Vt tiles: TDM tensor_load_to_lds with
// hardware row padding reproducing VT_STRIDE=40 (one descriptor per tile,
// issued by wave 0, TENSORcnt-synchronized).  Q fragments register-resident.
// ---------------------------------------------------------------------------
__global__ void __launch_bounds__(256) flash_attn(
    const _Float16* __restrict__ Qh, const _Float16* __restrict__ Kh,
    const _Float16* __restrict__ Vtg, float* __restrict__ Out)
{
    extern __shared__ __align__(16) char smem_raw[];
    _Float16* Ks0 = (_Float16*)smem_raw;            // 32 x 1032
    _Float16* Ks1 = Ks0 + 32 * QS_STRIDE;
    _Float16* Vt0 = Ks1 + 32 * QS_STRIDE;           // 1024 x 40
    _Float16* Vt1 = Vt0 + 1024 * VT_STRIDE;
    _Float16* Ps  = Vt1 + 1024 * VT_STRIDE;         // 32 x 40
    float*    Sp    = (float*)(Ps + 32 * PS_STRIDE); // 2 x 32 x 33
    float*    rmax  = Sp + 2 * 32 * 33;
    float*    rsum  = rmax + 32;
    float*    alpha = rsum + 32;

    const int t = threadIdx.x, lane = t & 31, w = t >> 5;
    const int b = blockIdx.y;
    const int qbase = blockIdx.x * 32;
    const size_t boff = (size_t)b * SEQ * D_MODEL;

    const int mi = w & 1, ni = (w >> 1) & 1, kh = w >> 2;  // S-phase role
    const int mo = w & 1, dch = w >> 1;                    // O ownership

    const _Float16* Kg = Kh + boff;
    const _Float16* Vg = Vtg + (size_t)b * D_MODEL * SEQ;  // [d][n]

    auto stage_kv = [&](_Float16* KsB, _Float16* VtB, int nb) {
        // K tile 32x1024: async b128, 16 per thread
#pragma unroll
        for (int i = 0; i < 16; ++i) {
            int h = t * 8 + i * 2048;
            int row = h >> 10, col = h & 1023;
            cp_b128(KsB + row * QS_STRIDE + col, Kg + (size_t)(nb + row) * D_MODEL + col);
        }
        // Vt tile 1024x32: one TDM descriptor (wave 0) or async fallback
#ifdef HAVE_TDM
        if (w == 0) {
            tdm_load_2d(lds_addr_of(VtB), (unsigned long long)(const void*)(Vg + nb),
                        /*tile*/ 32, 1024, /*tensor*/ SEQ, D_MODEL,
                        /*row_stride*/ SEQ,
                        /*pad_interval: 16 DW = 32 halves*/ 3,
                        /*pad_amount: 4 DW = 8 halves*/ 3, 1);
        }
#else
#pragma unroll
        for (int i = 0; i < 16; ++i) {
            int h = t * 8 + i * 2048;
            int d = h >> 5, cc = h & 31;
            cp_b128(VtB + d * VT_STRIDE + cc, Vg + (size_t)d * SEQ + nb + cc);
        }
#endif
    };

    if (t < 32) { rmax[t] = -1e30f; rsum[t] = 0.f; }

    stage_kv(Ks0, Vt0, 0);

    // loop-invariant Q A-fragments straight from global into registers
    v16h qf[16];
    const _Float16* Qg = Qh + boff + (size_t)qbase * D_MODEL;
#pragma unroll
    for (int kk = 0; kk < 16; ++kk)
        qf[kk] = ld_frag(Qg, mi * 16 + (lane & 15), D_MODEL, kh * 512 + kk * 32);

    v8f o[16];
#pragma unroll
    for (int i = 0; i < 16; ++i)
#pragma unroll
        for (int r = 0; r < 8; ++r) o[i][r] = 0.f;

    for (int it = 0; it < SEQ / 32; ++it) {
        const int nb = it * 32;
        _Float16* KsC = (it & 1) ? Ks1 : Ks0;
        _Float16* VtC = (it & 1) ? Vt1 : Vt0;

        wait_async();
#ifdef HAVE_TDM
        if (w == 0) wait_tensor();
#endif
        __syncthreads();                                    // tile (it) visible to all waves
        if (it + 1 < SEQ / 32)
            stage_kv((it & 1) ? Ks0 : Ks1, (it & 1) ? Vt0 : Vt1, nb + 32);

        // ---- S = Q K^T (quadrant (mi,ni), K-half kh) ----
        v8f s;
#pragma unroll
        for (int r = 0; r < 8; ++r) s[r] = 0.f;
#pragma unroll
        for (int kk = 0; kk < 16; ++kk) {
            v16h bk = ld_frag(KsC, ni * 16 + (lane & 15), QS_STRIDE, kh * 512 + kk * 32);
            s = __builtin_amdgcn_wmma_f32_16x16x32_f16(false, qf[kk], false, bk, (short)0, s, false, false);
        }
        {
            int cc = ni * 16 + (lane & 15);
            int rb = mi * 16 + ((lane >> 4) ? 8 : 0);
            float* dst = Sp + kh * (32 * 33);
#pragma unroll
            for (int r = 0; r < 8; ++r) dst[(rb + r) * 33 + cc] = s[r];
        }
        __syncthreads();

        // ---- online softmax: wave 0, one row per lane ----
        if (t < 32) {
            float sr[32];
            float m_old = rmax[t];
            float mx = m_old;
#pragma unroll
            for (int j = 0; j < 32; ++j) {
                float sv = (Sp[t * 33 + j] + Sp[32 * 33 + t * 33 + j]) * 0.03125f;
                sr[j] = sv;
                mx = fmaxf(mx, sv);
            }
            float al = __expf(m_old - mx);
            float sum = 0.f;
#pragma unroll
            for (int j = 0; j < 32; ++j) {
                float p = __expf(sr[j] - mx);
                Ps[t * PS_STRIDE + j] = (_Float16)p;
                sum += p;
            }
            rmax[t] = mx;
            rsum[t] = rsum[t] * al + sum;
            alpha[t] = al;
        }
        __syncthreads();

        // ---- rescale O, then O += P * V ----
        float sc[8];
        const int rb2 = mo * 16 + ((lane >> 4) ? 8 : 0);
#pragma unroll
        for (int r = 0; r < 8; ++r) sc[r] = alpha[rb2 + r];
#pragma unroll
        for (int i = 0; i < 16; ++i)
#pragma unroll
            for (int r = 0; r < 8; ++r) o[i][r] *= sc[r];

        v16h ap = ld_frag(Ps, mo * 16 + (lane & 15), PS_STRIDE, 0);
#pragma unroll
        for (int dt = 0; dt < 16; ++dt) {
            v16h bv = ld_frag(VtC, dch * 256 + dt * 16 + (lane & 15), VT_STRIDE, 0);
            o[dt] = __builtin_amdgcn_wmma_f32_16x16x32_f16(false, ap, false, bv, (short)0, o[dt], false, false);
        }
        // next iteration's top barrier protects KsC/VtC reuse
    }

    // ---- epilogue: O / l, fp32 store ----
    float inv[8];
    const int rb3 = mo * 16 + ((lane >> 4) ? 8 : 0);
#pragma unroll
    for (int r = 0; r < 8; ++r) inv[r] = 1.f / rsum[rb3 + r];
#pragma unroll
    for (int dt = 0; dt < 16; ++dt) {
        int col = dch * 256 + dt * 16 + (lane & 15);
#pragma unroll
        for (int r = 0; r < 8; ++r)
            Out[boff + (size_t)(qbase + rb3 + r) * D_MODEL + col] = o[dt][r] * inv[r];
    }
}

// ---------------------------------------------------------------------------
extern "C" void kernel_launch(void* const* d_in, const int* in_sizes, int n_in,
                              void* d_out, int out_size, void* d_ws, size_t ws_size,
                              hipStream_t stream) {
    (void)in_sizes; (void)n_in; (void)out_size; (void)ws_size;
    const float* x  = (const float*)d_in[0];
    const float* Wq = (const float*)d_in[1];
    const float* Wk = (const float*)d_in[2];
    const float* Wv = (const float*)d_in[3];
    float* out = (float*)d_out;

    char* ws = (char*)d_ws;
    size_t off = 0;
    auto carve = [&](size_t bytes) -> void* {
        void* p = ws + off;
        off = (off + bytes + 255) & ~(size_t)255;
        return p;
    };
    const size_t nx = (size_t)MTOT * D_MODEL;
    const size_t nw = (size_t)D_MODEL * D_MODEL;
    _Float16* xh  = (_Float16*)carve(nx * 2);
    _Float16* Wqh = (_Float16*)carve(nw * 2);
    _Float16* Wkh = (_Float16*)carve(nw * 2);
    _Float16* Wvh = (_Float16*)carve(nw * 2);
    _Float16* Qh  = (_Float16*)carve(nx * 2);
    _Float16* Kh  = (_Float16*)carve(nx * 2);
    _Float16* Vtg = (_Float16*)carve(nx * 2);

    cast_f32_to_f16<<<(int)((nx / 4 + 255) / 256), 256, 0, stream>>>(x,  xh,  (int)nx);
    cast_f32_to_f16<<<(int)((nw / 4 + 255) / 256), 256, 0, stream>>>(Wq, Wqh, (int)nw);
    cast_f32_to_f16<<<(int)((nw / 4 + 255) / 256), 256, 0, stream>>>(Wk, Wkh, (int)nw);
    cast_f32_to_f16<<<(int)((nw / 4 + 255) / 256), 256, 0, stream>>>(Wv, Wvh, (int)nw);

    qkv_gemm<<<dim3(MTOT / 64, 24), 256, 0, stream>>>(xh, Wqh, Wkh, Wvh, Qh, Kh, Vtg);

    const int smem_bytes = (2 * 32 * QS_STRIDE + 2 * 1024 * VT_STRIDE + 32 * PS_STRIDE) * 2
                         + (2 * 32 * 33 + 3 * 32) * 4;   // 307,328 B  (< 320 KB WGP LDS)
    flash_attn<<<dim3(SEQ / 32, BATCH), 256, smem_bytes, stream>>>(Qh, Kh, Vtg, out);
}
